// AttentionMechanism_79826262164169
// MI455X (gfx1250) — matrix operand order repaired and checked by reading, not compiled, not run on hardware
//
#include <hip/hip_runtime.h>
#include <math.h>

typedef __attribute__((ext_vector_type(16))) __bf16 v16bf;
typedef __attribute__((ext_vector_type(8)))  float  v8f;

#define DD 512
#define SS 8192
#define BB 8

// ---------------------------------------------------------------------------
// Kernel 1: q[b,e] = dot(query[b,:], Wq[e,:]) + bq[e]
// ---------------------------------------------------------------------------
__global__ __launch_bounds__(512) void qproj_kernel(
    const float* __restrict__ query, const float* __restrict__ Wq,
    const float* __restrict__ bq, float* __restrict__ qout) {
  const int b = blockIdx.x;
  const int e = threadIdx.x;
  __shared__ float qs[DD];
  qs[e] = query[b * DD + e];
  __syncthreads();
  const float* w = Wq + (size_t)e * DD;
  float acc = bq[e];
  for (int d = 0; d < DD; d += 4) {
    float4 wv = *(const float4*)(w + d);
    acc += wv.x * qs[d] + wv.y * qs[d + 1] + wv.z * qs[d + 2] + wv.w * qs[d + 3];
  }
  qout[b * DD + e] = acc;
}

// ---------------------------------------------------------------------------
// Kernel 2: fused keys-GEMM (bf16 WMMA, f32 accum) + scores = keys . q
// Workgroup: 256 threads = 8 waves. Tile: 32 S-rows x 512 E-cols, K=512.
//   wave = s_half(2) x e_grp(4): each wave owns 16 rows x 128 cols = 8 C tiles.
// A k-slice staged raw-f32 via GLOBAL_LOAD_ASYNC_TO_LDS (ASYNCcnt path),
// converted to bf16 at fragment-build time. B k-slice staged as bf16 via the
// VGPR path (one conversion per Wk element per workgroup; L2-resident).
// ---------------------------------------------------------------------------
__global__ __launch_bounds__(256) void keys_scores_kernel(
    const float* __restrict__ inputs, const float* __restrict__ Wk,
    const float* __restrict__ bk, const float* __restrict__ qproj,
    float* __restrict__ scores) {
  __shared__ float  Alds[32 * 32];    // A k-slice  (32 rows x 32 k), f32
  __shared__ __bf16 Blds[DD * 32];    // Wk k-slice (512 e  x 32 k), bf16
  __shared__ float  qlds[DD];
  __shared__ float  slds[32];

  const int b       = blockIdx.y;
  const int s_block = blockIdx.x * 32;
  const int tid     = threadIdx.x;
  const int lane    = tid & 31;
  const int wave    = tid >> 5;
  const int e_grp   = wave & 3;   // 128-col group
  const int s_half  = wave >> 2;  // 16-row group

  // stage q[b,:] once; zero score accumulators
  for (int i = tid; i < DD; i += 256) qlds[i] = qproj[b * DD + i];
  if (tid < 32) slds[tid] = 0.f;

  const float* Abase = inputs + ((size_t)b * SS + s_block) * DD;

  // per-thread async-copy slot: 16 bytes of the A slice
  const int aflat = tid * 4;                   // element index in 32x32 slice
  const int arow = aflat >> 5, acol = aflat & 31;
  const unsigned ldsA = (unsigned)(uintptr_t)(&Alds[aflat]);

  v8f acc[8] = {};

  __syncthreads();

  for (int kk = 0; kk < 16; ++kk) {
    const int k0 = kk * 32;
    // ---- stage A slice: global -> LDS direct (async, 16B per lane) ----
    {
      const float* src = Abase + (size_t)arow * DD + k0 + acol;
      asm volatile("global_load_async_to_lds_b128 %0, %1, off"
                   :: "v"(ldsA), "v"((unsigned long long)(uintptr_t)src)
                   : "memory");
      if (kk < 15) __builtin_prefetch(src + 32, 0, 1);  // next k-slice
    }
    // ---- stage B slice: Wk[0:512, k0:k0+32] f32 -> bf16 (64 / thread) ----
#pragma unroll
    for (int i = 0; i < 16; ++i) {
      const int flat = i * 1024 + tid * 4;
      const int e = flat >> 5, k = flat & 31;
      float4 v = *(const float4*)(Wk + (size_t)e * DD + k0 + k);
      __bf16* dst = &Blds[e * 32 + k];
      dst[0] = (__bf16)v.x; dst[1] = (__bf16)v.y;
      dst[2] = (__bf16)v.z; dst[3] = (__bf16)v.w;
    }
    asm volatile("s_wait_asynccnt 0x0" ::: "memory");
    __syncthreads();

    // ---- A fragment (ISA 16-bit 16x32 layout), convert f32->bf16 here ----
    // lanes 0-15: M=lane, halves 0..7 -> K 0..7, 8..15 -> K 16..23
    // lanes 16-31: M=lane-16, halves 0..7 -> K 8..15, 8..15 -> K 24..31
    v16bf afrag;
    {
      const int m  = lane & 15;
      const int kb = (lane >> 4) * 8;
      const float* ap = &Alds[(s_half * 16 + m) * 32];
#pragma unroll
      for (int h = 0; h < 8; ++h) afrag[h]     = (__bf16)ap[kb + h];
#pragma unroll
      for (int h = 0; h < 8; ++h) afrag[8 + h] = (__bf16)ap[16 + kb + h];
    }
    // ---- 8 N-tiles: B fragment (pipelined 1 tile ahead) + WMMA ----
    // B (32x16): lanes 0-15 hold col n=lane, K 0..7,16..23; lanes 16-31 K 8..15,24..31
    const int kb = (lane >> 4) * 8;
    const __bf16* bbase = &Blds[(e_grp * 128 + (lane & 15)) * 32 + kb];
    v16bf bf[2];
    ((uint4*)&bf[0])[0] = *(const uint4*)(bbase);
    ((uint4*)&bf[0])[1] = *(const uint4*)(bbase + 16);
#pragma unroll
    for (int t = 0; t < 8; ++t) {
      if (t < 7) {
        const __bf16* bp = bbase + (t + 1) * 16 * 32;
        ((uint4*)&bf[(t + 1) & 1])[0] = *(const uint4*)(bp);
        ((uint4*)&bf[(t + 1) & 1])[1] = *(const uint4*)(bp + 16);
      }
      acc[t] = __builtin_amdgcn_wmma_f32_16x16x32_bf16(
          false, afrag, false, bf[t & 1], (short)0, acc[t], false, false);
    }
    __syncthreads();
  }

  // ---- epilogue: scores[s] = sum_e (keys[s,e] + bk[e]) * q[b,e] ----
  // C layout: VGPR v, lanes 0-15: M=v,   N=lane
  //                    lanes16-31: M=v+8, N=lane-16
  float p[8];
  for (int v = 0; v < 8; ++v) p[v] = 0.f;
#pragma unroll
  for (int t = 0; t < 8; ++t) {
    const int n_abs = e_grp * 128 + t * 16 + (lane & 15);
    const float qv  = qlds[n_abs];
    const float bkv = bk[n_abs];
#pragma unroll
    for (int v = 0; v < 8; ++v) p[v] += (acc[t][v] + bkv) * qv;
  }
  const int m_base = s_half * 16 + (lane >> 4) * 8;
  for (int v = 0; v < 8; ++v) atomicAdd(&slds[m_base + v], p[v]);
  __syncthreads();
  if (tid < 32) scores[(size_t)b * SS + s_block + tid] = slds[tid];
}

// ---------------------------------------------------------------------------
// Kernel 3: softmax over S per batch; writes attn; zeroes pooled scratch
// ---------------------------------------------------------------------------
__global__ __launch_bounds__(1024) void softmax_kernel(
    const float* __restrict__ scores, float* __restrict__ attn,
    float* __restrict__ pooled) {
  const int b = blockIdx.x, tid = threadIdx.x;
  __shared__ float red[1024];
  float v[8];
  float m = -INFINITY;
  for (int i = 0; i < 8; ++i) {
    v[i] = scores[(size_t)b * SS + tid + i * 1024];
    m = fmaxf(m, v[i]);
  }
  red[tid] = m;
  __syncthreads();
  for (int off = 512; off > 0; off >>= 1) {
    if (tid < off) red[tid] = fmaxf(red[tid], red[tid + off]);
    __syncthreads();
  }
  m = red[0];
  __syncthreads();
  float s = 0.f;
  for (int i = 0; i < 8; ++i) { v[i] = __expf(v[i] - m); s += v[i]; }
  red[tid] = s;
  __syncthreads();
  for (int off = 512; off > 0; off >>= 1) {
    if (tid < off) red[tid] += red[tid + off];
    __syncthreads();
  }
  const float inv = 1.0f / red[0];
  for (int i = 0; i < 8; ++i)
    attn[(size_t)b * SS + tid + i * 1024] = v[i] * inv;
  if (tid < DD) pooled[b * DD + tid] = 0.f;
}

// ---------------------------------------------------------------------------
// Kernel 4: pooled[b,d] = sum_s attn[b,s] * inputs[b,s,d]   (BW-bound, L2-hot)
// ---------------------------------------------------------------------------
__global__ __launch_bounds__(512) void pool_kernel(
    const float* __restrict__ inputs, const float* __restrict__ attn,
    float* __restrict__ pooled) {
  const int b = blockIdx.y;
  const int chunk = blockIdx.x;  // 64 chunks of 128 rows
  const int d = threadIdx.x;
  const float* base = inputs + ((size_t)b * SS + chunk * 128) * DD;
  const float* a    = attn + (size_t)b * SS + chunk * 128;
  float acc = 0.f;
  for (int s = 0; s < 128; ++s) acc += a[s] * base[(size_t)s * DD + d];
#if defined(__HIP_DEVICE_COMPILE__)
  unsafeAtomicAdd(&pooled[b * DD + d], acc);
#endif
}

// ---------------------------------------------------------------------------
// Kernel 5: out[b,e] = dot(pooled[b,:], Wk[e,:]) + bk[e]
//   (valid since out = attn@keys = (attn@inputs)@Wk^T + bk, sum(attn)=1)
// ---------------------------------------------------------------------------
__global__ __launch_bounds__(512) void out_kernel(
    const float* __restrict__ Wk, const float* __restrict__ bk,
    const float* __restrict__ pooled, float* __restrict__ out) {
  const int b = blockIdx.x;
  const int e = threadIdx.x;
  __shared__ float ps[DD];
  ps[e] = pooled[b * DD + e];
  __syncthreads();
  const float* w = Wk + (size_t)e * DD;
  float acc = bk[e];
  for (int d = 0; d < DD; d += 4) {
    float4 wv = *(const float4*)(w + d);
    acc += wv.x * ps[d] + wv.y * ps[d + 1] + wv.z * ps[d + 2] + wv.w * ps[d + 3];
  }
  out[b * DD + e] = acc;
}

// ---------------------------------------------------------------------------
extern "C" void kernel_launch(void* const* d_in, const int* in_sizes, int n_in,
                              void* d_out, int out_size, void* d_ws, size_t ws_size,
                              hipStream_t stream) {
  const float* inputs = (const float*)d_in[0];  // [8,8192,512]
  const float* query  = (const float*)d_in[1];  // [8,512]
  const float* Wq     = (const float*)d_in[2];  // [512,512]
  const float* bq     = (const float*)d_in[3];  // [512]
  const float* Wk     = (const float*)d_in[4];  // [512,512]
  const float* bk     = (const float*)d_in[5];  // [512]

  float* out  = (float*)d_out;       // [8,512]
  float* attn = out + BB * DD;       // [8,8192]

  float* q      = (float*)d_ws;      // 4096 floats
  float* scores = q + BB * DD;       // 65536 floats
  float* pooled = scores + BB * SS;  // 4096 floats

  qproj_kernel<<<dim3(BB), dim3(DD), 0, stream>>>(query, Wq, bq, q);
  keys_scores_kernel<<<dim3(SS / 32, BB), dim3(256), 0, stream>>>(
      inputs, Wk, bk, q, scores);
  softmax_kernel<<<dim3(BB), dim3(1024), 0, stream>>>(scores, attn, pooled);
  pool_kernel<<<dim3(64, BB), dim3(DD), 0, stream>>>(inputs, attn, pooled);
  out_kernel<<<dim3(BB), dim3(DD), 0, stream>>>(Wk, bk, pooled, out);
}
